// DeformConvReLU62d_15719580304166
// MI455X (gfx1250) — compile-verified, matrix operand order checked by reading, not run on hardware
//
#include <hip/hip_runtime.h>
#include <cstdint>

typedef __bf16  v16bf __attribute__((ext_vector_type(16)));
typedef float   v8f   __attribute__((ext_vector_type(8)));

#define CIN    64
#define COUT   64
#define HW     16384          // 128*128
#define KTAPS  9
#define KC     576            // CIN*KTAPS (reduction dim, ordered k*64+c)
#define NPIX   128            // output pixels per block (N tile)
#define THREADS 256

__device__ __forceinline__ unsigned short f2bf(float f) {
  unsigned u = __builtin_bit_cast(unsigned, f);
  u += 0x7FFFu + ((u >> 16) & 1u);          // round-to-nearest-even
  return (unsigned short)(u >> 16);
}

// Reorder + convert weight [Cout][Cin][3][3] f32 -> bf16 [Cout][k*64+c]
__global__ void dcn_prep_weight_bf16(const float* __restrict__ w,
                                     unsigned short* __restrict__ wb) {
  int idx = blockIdx.x * THREADS + threadIdx.x;
  if (idx >= COUT * KC) return;
  int o = idx / KC;
  int r = idx - o * KC;
  int k = r >> 6;          // tap
  int c = r & 63;          // input channel
  wb[idx] = f2bf(w[(o * CIN + c) * KTAPS + k]);
}

__global__ __launch_bounds__(THREADS, 1)
void dcn_wmma_kernel(const float* __restrict__ x,
                     const float* __restrict__ off,
                     const float* __restrict__ msk,
                     const unsigned short* __restrict__ wb,
                     const float* __restrict__ bias,
                     float* __restrict__ out) {
  extern __shared__ char smem[];
  unsigned short* Wlds = (unsigned short*)smem;                    // [COUT][KC] bf16, 72KB
  unsigned short* Slds = (unsigned short*)(smem + COUT * KC * 2);  // [NPIX][KC] bf16, 144KB

  const int tid    = threadIdx.x;
  const int b      = blockIdx.x >> 7;            // 8 batches * 128 tiles
  const int p_base = (blockIdx.x & 127) * NPIX;

  // ---- Phase 1a: async DMA of bf16 weight tile global -> LDS (ASYNCcnt path) ----
  {
    unsigned wlds_base = (unsigned)(uintptr_t)(void*)Wlds;  // flat LDS addr low 32b == LDS offset
    for (int i = tid; i < (COUT * KC * 2) / 16; i += THREADS) {
      unsigned byteoff = (unsigned)i * 16u;
      unsigned ldst    = wlds_base + byteoff;
      asm volatile("global_load_async_to_lds_b128 %0, %1, %2"
                   :: "v"(ldst), "v"(byteoff), "s"(wb) : "memory");
    }
  }

  // ---- Phase 1b: deformable im2col (bilinear + mask) -> LDS bf16 ----
  const float* xb = x + (size_t)b * CIN * HW;
  for (int t = tid; t < NPIX * KTAPS; t += THREADS) {
    int nl = t / KTAPS;                 // local pixel
    int k  = t - nl * KTAPS;            // tap
    int p  = p_base + nl;
    int h  = p >> 7;
    int w_ = p & 127;
    int ky = k / 3, kx = k - ky * 3;

    float offy = off[(size_t)(b * 18 + 2 * k    ) * HW + p];
    float offx = off[(size_t)(b * 18 + 2 * k + 1) * HW + p];
    float mv   = msk[(size_t)(b * KTAPS + k) * HW + p];

    float py = (float)(h - 1 + ky) + offy;   // stride 1, pad 1, dil 1
    float px = (float)(w_ - 1 + kx) + offx;
    float y0f = floorf(py), x0f = floorf(px);
    float wy = py - y0f,    wx = px - x0f;
    int y0 = (int)y0f, x0 = (int)x0f;
    int y1 = y0 + 1,   x1 = x0 + 1;

    float vy0 = (y0 >= 0 && y0 < 128) ? 1.f : 0.f;
    float vy1 = (y1 >= 0 && y1 < 128) ? 1.f : 0.f;
    float vx0 = (x0 >= 0 && x0 < 128) ? 1.f : 0.f;
    float vx1 = (x1 >= 0 && x1 < 128) ? 1.f : 0.f;
    // fold mask and validity into the 4 tap weights -> branch-free channel loop
    float w00 = (1.f - wy) * (1.f - wx) * mv * vy0 * vx0;
    float w01 = (1.f - wy) * wx         * mv * vy0 * vx1;
    float w10 = wy * (1.f - wx)         * mv * vy1 * vx0;
    float w11 = wy * wx                 * mv * vy1 * vx1;

    int y0c = min(max(y0, 0), 127), y1c = min(max(y1, 0), 127);
    int x0c = min(max(x0, 0), 127), x1c = min(max(x1, 0), 127);
    int i00 = y0c * 128 + x0c, i01 = y0c * 128 + x1c;
    int i10 = y1c * 128 + x0c, i11 = y1c * 128 + x1c;

    const float* xp = xb;
    unsigned* srow32 = (unsigned*)(Slds + nl * KC + k * CIN);  // 4B aligned
#pragma unroll 4
    for (int c = 0; c < CIN; c += 2) {
      float v0 = fmaf(xp[i00], w00,
                 fmaf(xp[i01], w01,
                 fmaf(xp[i10], w10, xp[i11] * w11)));
      const float* xq = xp + HW;
      float v1 = fmaf(xq[i00], w00,
                 fmaf(xq[i01], w01,
                 fmaf(xq[i10], w10, xq[i11] * w11)));
      srow32[c >> 1] = (unsigned)f2bf(v0) | ((unsigned)f2bf(v1) << 16);
      xp += 2 * HW;
    }
  }

  asm volatile("s_wait_asynccnt 0" ::: "memory");
  __syncthreads();

  // ---- Phase 2: WMMA  out[64 x 128] = W[64 x 576] x S^T[576 x 128] ----
  const int lane   = tid & 31;
  const int wave   = tid >> 5;
  const int wave_m = wave & 3;      // 4 waves over M (Cout)
  const int wave_n = wave >> 2;     // 2 waves over N, 4 n-tiles of 16 each
  const int col    = lane & 15;
  const int khalf  = (lane >> 4) << 4;   // lanes 16-31 carry K+16..K+31
  const int mo     = (lane >> 4) * 8;    // C/D: lanes 16-31 hold M=8..15 rows

  v8f acc[4];
#pragma unroll
  for (int r = 0; r < 8; ++r) {
    float bv = bias[wave_m * 16 + mo + r];
    acc[0][r] = bv; acc[1][r] = bv; acc[2][r] = bv; acc[3][r] = bv;
  }

  // A layout used here (contiguous k, B same): lane%16=row M, VGPR0..7 pack K=khalf..khalf+15
  const unsigned short* abase = Wlds + (wave_m * 16 + col) * KC + khalf;
  const unsigned short* bbase[4];
#pragma unroll
  for (int j = 0; j < 4; ++j)
    bbase[j] = Slds + (wave_n * 64 + j * 16 + col) * KC + khalf;

#pragma unroll 2
  for (int kk = 0; kk < KC / 32; ++kk) {
    int k0 = kk * 32;
    v16bf a = *(const v16bf*)(abase + k0);   // 32B aligned: rows are 1152B
#pragma unroll
    for (int j = 0; j < 4; ++j) {
      v16bf bfrag = *(const v16bf*)(bbase[j] + k0);
      acc[j] = __builtin_amdgcn_wmma_f32_16x16x32_bf16(false, a, false, bfrag,
                                                       (short)0, acc[j], false, false);
    }
  }

  // ---- Epilogue: ReLU6 + scatter to [B, Cout, Ho, Wo] ----
#pragma unroll
  for (int j = 0; j < 4; ++j) {
    int n = wave_n * 64 + j * 16 + col;
    int p = p_base + n;
    float* op = out + (size_t)(b * COUT + wave_m * 16 + mo) * HW + p;
#pragma unroll
    for (int r = 0; r < 8; ++r) {
      float v = acc[j][r];
      v = fminf(fmaxf(v, 0.f), 6.f);
      op[(size_t)r * HW] = v;
    }
  }
}

extern "C" void kernel_launch(void* const* d_in, const int* in_sizes, int n_in,
                              void* d_out, int out_size, void* d_ws, size_t ws_size,
                              hipStream_t stream) {
  const float* x    = (const float*)d_in[0];   // [8,64,128,128]
  const float* off  = (const float*)d_in[1];   // [8,18,128,128]
  const float* msk  = (const float*)d_in[2];   // [8,9,128,128]
  const float* wgt  = (const float*)d_in[3];   // [64,64,3,3]
  const float* bias = (const float*)d_in[4];   // [64]
  float* out = (float*)d_out;                  // [8,64,128,128]

  unsigned short* wb = (unsigned short*)d_ws;  // bf16 weight, 73728 bytes

  dcn_prep_weight_bf16<<<(COUT * KC + THREADS - 1) / THREADS, THREADS, 0, stream>>>(wgt, wb);

  const int nblocks = (8 * HW) / NPIX;         // 1024
  const size_t lds_bytes = (size_t)(COUT * KC + NPIX * KC) * 2;  // 221184 B (<320KB WGP)
  dcn_wmma_kernel<<<nblocks, THREADS, lds_bytes, stream>>>(x, off, msk, wb, bias, out);
}
